// GlobalInteractor_35519379538325
// MI455X (gfx1250) — matrix operand order repaired and checked by reading, not compiled, not run on hardware
//
#include <hip/hip_runtime.h>
#include <hip/hip_bf16.h>

// ---------------------------------------------------------------------------
// GlobalInteractor (graph attention, 3 layers) for MI455X / gfx1250, wave32.
// All D=64 projections run on v_wmma_f32_16x16x32_bf16 (bf16 A/B, f32 acc).
// Node-side projections hoisted from E (1M edges) to N (64K nodes): 16x less
// GEMM work + bytes than the naive edge-side formulation. Edge-side ke/ve
// GEMMs are fused into the score/aggregate passes (never materialized).
// Weights are pre-packed once into WMMA B-fragment order (bf16) so hot-loop
// fragment loads are two 16B vector loads, no per-use convert.
// ---------------------------------------------------------------------------

#define NN 65536
#define TT 20
#define EE 1048576
#define DD 64
#define HH 8
#define LL 3
#define HSM1 19

typedef __attribute__((ext_vector_type(16))) __bf16        v16bf;
typedef __attribute__((ext_vector_type(8)))  float         v8f;
typedef __attribute__((ext_vector_type(4)))  unsigned int  v4u;

union AFrag { v4u u[2]; v16bf v; };
union BFrag { unsigned short s[16]; v16bf v; };

__device__ __forceinline__ unsigned short f2bf(float f) {
    unsigned int u = __float_as_uint(f);
    unsigned int r = (u + 0x7FFFu + ((u >> 16) & 1u)) >> 16;  // RNE
    return (unsigned short)r;
}

__device__ __forceinline__ v8f wmma_bf16(v16bf a, v16bf b, v8f c) {
    return __builtin_amdgcn_wmma_f32_16x16x32_bf16(false, a, false, b,
                                                   (short)0, c, false, false);
}

__device__ __forceinline__ void wait_ds() {
    asm volatile("s_wait_dscnt 0" ::: "memory");
}

// A fragment (16x32 bf16 tile, K block kb) from row-major bf16 [rows, ld].
// Lane L<16 holds row L: elems 0..7 = K 0..7, elems 8..15 = K 16..23 (+kb*32).
// Lane L>=16 holds row L-16: K 8..15 and 24..31. Two 16B loads per lane.
__device__ __forceinline__ v16bf load_a_bf16(const unsigned short* base,
                                             long row0, int ld, int kb, int lane) {
    int half = (lane >> 4) & 1, r = lane & 15;
    const unsigned short* p = base + (size_t)(row0 + r) * ld + kb * 32 + 8 * half;
    AFrag f;
    f.u[0] = *(const v4u*)(const void*)p;
    f.u[1] = *(const v4u*)(const void*)(p + 16);
    return f.v;
}

// Same A fragment but from row-major f32 with on-the-fly bf16 convert.
__device__ __forceinline__ v16bf load_a_f32(const float* base, long row0,
                                            int ld, int kb, int lane) {
    int half = (lane >> 4) & 1, r = lane & 15;
    const float* p = base + (size_t)(row0 + r) * ld + kb * 32 + 8 * half;
    BFrag f;
#pragma unroll
    for (int e = 0; e < 8; ++e) {
        f.s[e]     = f2bf(p[e]);
        f.s[8 + e] = f2bf(p[16 + e]);
    }
    return f.v;
}

// B fragment from pre-packed bf16 weights. Packed block index = kb*nbt + nt;
// each block is 32 lanes x 16 bf16 (32B per lane, contiguous).
__device__ __forceinline__ v16bf load_b_packed(const unsigned short* P, int nbt,
                                               int nt, int kb, int lane) {
    const unsigned short* p = P + ((size_t)(kb * nbt + nt) * 32 + lane) * 16;
    AFrag f;
    f.u[0] = *(const v4u*)(const void*)p;
    f.u[1] = *(const v4u*)(const void*)(p + 16);
    return f.v;
}

__device__ __forceinline__ void atomicMaxFloat(float* addr, float val) {
    if (val >= 0.f) atomicMax((int*)addr, __float_as_int(val));
    else            atomicMin((unsigned int*)addr, (unsigned int)__float_as_int(val));
}

// ---------------------------------------------------------------------------
// 0) Weight pre-pack: row-major f32 W[K, Ncols] -> B-fragment-ordered bf16.
//    Fragment (kb, nt): lane holds column (lane&15)+nt*16, elems e map to
//    K = kb*32 + 8*(lane>>4) + (e&7) + 16*(e>>3). One wave per fragment block.
// ---------------------------------------------------------------------------
__global__ void prepack_b_kernel(const float* __restrict__ W, int K, int Ncols,
                                 unsigned short* __restrict__ out)
{
    int lane = threadIdx.x & 31;
    int wid = (blockIdx.x * blockDim.x + threadIdx.x) >> 5;
    int nw = (gridDim.x * blockDim.x) >> 5;
    int nbt = Ncols >> 4;
    int nblocks = (K >> 5) * nbt;
    int half = (lane >> 4) & 1;
    for (int bidx = wid; bidx < nblocks; bidx += nw) {
        int kb = bidx / nbt, nt = bidx % nbt;
        int c = (lane & 15) + nt * 16;
        unsigned short* o = out + ((size_t)bidx * 32 + lane) * 16;
#pragma unroll
        for (int e = 0; e < 16; ++e)
            o[e] = f2bf(W[(size_t)(kb * 32 + 8 * half + (e & 7) + ((e >> 3) << 4)) * Ncols + c]);
    }
}

// ---------------------------------------------------------------------------
// 1) Per-edge relative-feature MLP: two branches (2->32, LN, ReLU, 32->64),
//    sum (+both biases), LN+ReLU -> u [E,64] bf16. Branch weights staged in
//    LDS (16KB) so the inner 32x64 products read shared memory.
// ---------------------------------------------------------------------------
__global__ void rel_phase1_kernel(
    const float* __restrict__ pos, const float* __restrict__ ang,
    const int* __restrict__ eidx,
    const float* __restrict__ r1w0, const float* __restrict__ r1b0,
    const float* __restrict__ r1g0, const float* __restrict__ r1bb0,
    const float* __restrict__ r1w1, const float* __restrict__ r1b1,
    const float* __restrict__ r2w0, const float* __restrict__ r2b0,
    const float* __restrict__ r2g0, const float* __restrict__ r2bb0,
    const float* __restrict__ r2w1, const float* __restrict__ r2b1,
    const float* __restrict__ ag0, const float* __restrict__ ab0,
    unsigned short* __restrict__ u)
{
    __shared__ float sw1[2][2048];   // 32x64 per branch
    __shared__ float sml[2][160];    // w0(64) | b0(32) | g0(32) | bb0(32)
    __shared__ float sb1[2][64];
    __shared__ float sag[128];       // ag0 | ab0

    for (int i = threadIdx.x; i < 2048; i += blockDim.x) {
        sw1[0][i] = r1w1[i]; sw1[1][i] = r2w1[i];
    }
    for (int i = threadIdx.x; i < 64; i += blockDim.x) {
        sml[0][i] = r1w0[i]; sml[1][i] = r2w0[i];
        sb1[0][i] = r1b1[i]; sb1[1][i] = r2b1[i];
        sag[i] = ag0[i]; sag[64 + i] = ab0[i];
    }
    for (int i = threadIdx.x; i < 32; i += blockDim.x) {
        sml[0][64 + i] = r1b0[i]; sml[0][96 + i] = r1g0[i]; sml[0][128 + i] = r1bb0[i];
        sml[1][64 + i] = r2b0[i]; sml[1][96 + i] = r2g0[i]; sml[1][128 + i] = r2bb0[i];
    }
    __syncthreads();

    int e = blockIdx.x * blockDim.x + threadIdx.x;
    if (e >= EE) return;
    int s = eidx[e], d = eidx[EE + e];
    float px = pos[(size_t)s * TT * 2 + HSM1 * 2]     - pos[(size_t)d * TT * 2 + HSM1 * 2];
    float py = pos[(size_t)s * TT * 2 + HSM1 * 2 + 1] - pos[(size_t)d * TT * 2 + HSM1 * 2 + 1];
    float ad = ang[d], asr = ang[s];
    float c = cosf(ad), sn = sinf(ad);
    float xin[2][2];
    xin[0][0] = px * c + py * sn;
    xin[0][1] = -px * sn + py * c;
    float th = asr - ad;
    xin[1][0] = cosf(th);
    xin[1][1] = sinf(th);

    float h[64];
#pragma unroll
    for (int k = 0; k < 64; ++k) h[k] = sb1[0][k] + sb1[1][k];  // both branch biases

#pragma unroll
    for (int br = 0; br < 2; ++br) {
        const float* w0 = sml[br];
        const float* b0 = sml[br] + 64;
        const float* g0 = sml[br] + 96;
        const float* bb0 = sml[br] + 128;
        const float* w1 = sw1[br];
        float xa = xin[br][0], xb = xin[br][1];
        float t[32], mu = 0.f;
#pragma unroll
        for (int j = 0; j < 32; ++j) { t[j] = xa * w0[j] + xb * w0[32 + j] + b0[j]; mu += t[j]; }
        mu *= (1.f / 32.f);
        float var = 0.f;
#pragma unroll
        for (int j = 0; j < 32; ++j) { float dv = t[j] - mu; var += dv * dv; }
        float rs = rsqrtf(var * (1.f / 32.f) + 1e-5f);
        for (int j = 0; j < 32; ++j) {
            float tn = fmaxf((t[j] - mu) * rs * g0[j] + bb0[j], 0.f);
#pragma unroll
            for (int k = 0; k < 64; ++k) h[k] += tn * w1[j * 64 + k];
        }
    }

    // LN(a_g0, a_b0) + ReLU -> u bf16
    float mu = 0.f;
#pragma unroll
    for (int k = 0; k < 64; ++k) mu += h[k];
    mu *= (1.f / 64.f);
    float var = 0.f;
#pragma unroll
    for (int k = 0; k < 64; ++k) { float dv = h[k] - mu; var += dv * dv; }
    float rs = rsqrtf(var * (1.f / 64.f) + 1e-5f);
#pragma unroll
    for (int k = 0; k < 64; ++k) {
        float v = fmaxf((h[k] - mu) * rs * sag[k] + sag[64 + k], 0.f);
        u[(size_t)e * 64 + k] = f2bf(v);
    }
}

// ---------------------------------------------------------------------------
// 2) WMMA GEMM (rows x 64) @ (64 x 64) + bias, then row LayerNorm -> bf16.
//    One wave per 16-row tile; LDS bounce for the cross-lane LN.
// ---------------------------------------------------------------------------
__global__ void gemm_ln_kernel(const unsigned short* __restrict__ A,
                               const unsigned short* __restrict__ Wp,
                               const float* __restrict__ bias,
                               const float* __restrict__ g,
                               const float* __restrict__ b,
                               unsigned short* __restrict__ out, int rows)
{
    __shared__ float lds[4][16][68];
    __shared__ float stats[4][16][2];
    int lane = threadIdx.x & 31;
    int wv = threadIdx.x >> 5;
    int wid = (blockIdx.x * blockDim.x + threadIdx.x) >> 5;
    int nw = (gridDim.x * blockDim.x) >> 5;
    int half = (lane >> 4) & 1;

    int ntiles = rows >> 4;
    for (int t = wid; t < ntiles; t += nw) {
        v16bf a0 = load_a_bf16(A, (long)t * 16, 64, 0, lane);
        v16bf a1 = load_a_bf16(A, (long)t * 16, 64, 1, lane);
#pragma unroll
        for (int nt = 0; nt < 4; ++nt) {
            v8f acc = {};
            acc = wmma_bf16(a0, load_b_packed(Wp, 4, nt, 0, lane), acc);
            acc = wmma_bf16(a1, load_b_packed(Wp, 4, nt, 1, lane), acc);
            int n = nt * 16 + (lane & 15);
            float bv = bias[n];
#pragma unroll
            for (int rr = 0; rr < 8; ++rr)
                lds[wv][rr + 8 * half][n] = acc[rr] + bv;
        }
        wait_ds();
        if (lane < 16) {
            float mu = 0.f;
            for (int k = 0; k < 64; ++k) mu += lds[wv][lane][k];
            mu *= (1.f / 64.f);
            float var = 0.f;
            for (int k = 0; k < 64; ++k) { float dv = lds[wv][lane][k] - mu; var += dv * dv; }
            stats[wv][lane][0] = mu;
            stats[wv][lane][1] = rsqrtf(var * (1.f / 64.f) + 1e-5f);
        }
        wait_ds();
#pragma unroll
        for (int i = 0; i < 32; ++i) {
            int idx = lane * 32 + i;
            int rrow = idx >> 6, cc = idx & 63;
            float mu = stats[wv][rrow][0], rs = stats[wv][rrow][1];
            float v = (lds[wv][rrow][cc] - mu) * rs * g[cc] + b[cc];
            out[(size_t)(t * 16 + rrow) * 64 + cc] = f2bf(v);
        }
        wait_ds();
    }
}

// ---------------------------------------------------------------------------
// 3) Row LayerNorm (f32 in -> bf16 out), one thread per 64-wide row.
// ---------------------------------------------------------------------------
__global__ void ln64_kernel(const float* __restrict__ x,
                            const float* __restrict__ g,
                            const float* __restrict__ b,
                            unsigned short* __restrict__ out, int rows)
{
    int r = blockIdx.x * blockDim.x + threadIdx.x;
    if (r >= rows) return;
    const float* p = x + (size_t)r * 64;
    float mu = 0.f;
#pragma unroll
    for (int k = 0; k < 64; ++k) mu += p[k];
    mu *= (1.f / 64.f);
    float var = 0.f;
#pragma unroll
    for (int k = 0; k < 64; ++k) { float dv = p[k] - mu; var += dv * dv; }
    float rs = rsqrtf(var * (1.f / 64.f) + 1e-5f);
#pragma unroll
    for (int k = 0; k < 64; ++k)
        out[(size_t)r * 64 + k] = f2bf((p[k] - mu) * rs * g[k] + b[k]);
}

// ---------------------------------------------------------------------------
// 4) Node projections Q/K/V = xn @ W + b  (3x WMMA GEMM, N x 64 x 64).
// ---------------------------------------------------------------------------
__device__ __forceinline__ void gemm_store_tile(v16bf a0, v16bf a1,
                                                const unsigned short* Wp,
                                                const float* bias,
                                                float* out, int t, int lane)
{
    int half = (lane >> 4) & 1;
#pragma unroll
    for (int nt = 0; nt < 4; ++nt) {
        v8f acc = {};
        acc = wmma_bf16(a0, load_b_packed(Wp, 4, nt, 0, lane), acc);
        acc = wmma_bf16(a1, load_b_packed(Wp, 4, nt, 1, lane), acc);
        int n = nt * 16 + (lane & 15);
        float bv = bias[n];
#pragma unroll
        for (int rr = 0; rr < 8; ++rr)
            out[(size_t)(t * 16 + rr + 8 * half) * 64 + n] = acc[rr] + bv;
    }
}

__global__ void qkv_kernel(const unsigned short* __restrict__ xn,
                           const unsigned short* __restrict__ Wqp, const float* __restrict__ bq,
                           const unsigned short* __restrict__ Wkp, const float* __restrict__ bk,
                           const unsigned short* __restrict__ Wvp, const float* __restrict__ bv,
                           float* __restrict__ Q, float* __restrict__ K, float* __restrict__ V)
{
    int lane = threadIdx.x & 31;
    int wid = (blockIdx.x * blockDim.x + threadIdx.x) >> 5;
    int nw = (gridDim.x * blockDim.x) >> 5;
    for (int t = wid; t < (NN >> 4); t += nw) {
        v16bf a0 = load_a_bf16(xn, (long)t * 16, 64, 0, lane);
        v16bf a1 = load_a_bf16(xn, (long)t * 16, 64, 1, lane);
        gemm_store_tile(a0, a1, Wqp, bq, Q, t, lane);
        gemm_store_tile(a0, a1, Wkp, bk, K, t, lane);
        gemm_store_tile(a0, a1, Wvp, bv, V, t, lane);
    }
}

// ---------------------------------------------------------------------------
// 5) init m=-1e30, denom=0, agg=0
// ---------------------------------------------------------------------------
__global__ void init_kernel(float* __restrict__ m, float* __restrict__ denom,
                            float* __restrict__ agg)
{
    int tid = blockIdx.x * blockDim.x + threadIdx.x;  // N*64 threads
    if (tid < NN * DD) agg[tid] = 0.f;
    if (tid < NN * HH) { m[tid] = -1e30f; denom[tid] = 0.f; }
}

// ---------------------------------------------------------------------------
// 6) Pass 1: ke = rel_e @ Wke (fused WMMA), score = q.(kn+ke)/sqrt(8),
//    segment max via float atomic-max. Head reduction via shfl_xor over the
//    8 lanes that hold one head's features.
// ---------------------------------------------------------------------------
__global__ void attn_score_kernel(const unsigned short* __restrict__ relE,
                                  const unsigned short* __restrict__ Wkep,
                                  const float* __restrict__ bke,
                                  const float* __restrict__ Qn, const float* __restrict__ Kn,
                                  const int* __restrict__ eidx,
                                  const unsigned char* __restrict__ pad,
                                  float* __restrict__ scores, float* __restrict__ mbuf)
{
    int lane = threadIdx.x & 31;
    int wid = (blockIdx.x * blockDim.x + threadIdx.x) >> 5;
    int nw = (gridDim.x * blockDim.x) >> 5;
    int half = (lane >> 4) & 1;

    for (int t = wid; t < (EE >> 4); t += nw) {
        v16bf a0 = load_a_bf16(relE, (long)t * 16, 64, 0, lane);
        v16bf a1 = load_a_bf16(relE, (long)t * 16, 64, 1, lane);
        int es[8], ds[8]; bool vl[8];
#pragma unroll
        for (int rr = 0; rr < 8; ++rr) {
            int e = t * 16 + rr + 8 * half;
            es[rr] = eidx[e]; ds[rr] = eidx[EE + e];
            vl[rr] = (pad[(size_t)es[rr] * TT + HSM1] == 0) &&
                     (pad[(size_t)ds[rr] * TT + HSM1] == 0);
        }
#pragma unroll
        for (int nt = 0; nt < 4; ++nt) {
            v8f acc = {};
            acc = wmma_bf16(a0, load_b_packed(Wkep, 4, nt, 0, lane), acc);
            acc = wmma_bf16(a1, load_b_packed(Wkep, 4, nt, 1, lane), acc);
            int n = nt * 16 + (lane & 15);
            float bb = bke[n];
            int h = nt * 2 + ((lane >> 3) & 1);
#pragma unroll
            for (int rr = 0; rr < 8; ++rr) {
                float q  = Qn[(size_t)ds[rr] * 64 + n];
                float kn = Kn[(size_t)es[rr] * 64 + n];
                float p = q * (kn + acc[rr] + bb);
                p += __shfl_xor(p, 1, 32);
                p += __shfl_xor(p, 2, 32);
                p += __shfl_xor(p, 4, 32);
                float sc = vl[rr] ? p * 0.35355339f : -1e9f;
                if ((lane & 7) == 0) {
                    int e = t * 16 + rr + 8 * half;
                    scores[(size_t)e * 8 + h] = sc;
                    atomicMaxFloat(&mbuf[(size_t)ds[rr] * 8 + h], sc);
                }
            }
        }
    }
}

// ---------------------------------------------------------------------------
// 7) Pass 2: e = exp(score - m[dst]); denom += e  (segment sum).
// ---------------------------------------------------------------------------
__global__ void attn_exp_kernel(float* __restrict__ scores,
                                const float* __restrict__ mbuf,
                                float* __restrict__ denom,
                                const int* __restrict__ eidx)
{
    size_t tid = (size_t)blockIdx.x * blockDim.x + threadIdx.x;
    if (tid >= (size_t)EE * 8) return;
    int h = (int)(tid & 7);
    size_t e = tid >> 3;
    int d = eidx[EE + e];
    float sc = scores[tid];
    float ex = (sc <= -1e8f) ? 0.f : __expf(sc - mbuf[(size_t)d * 8 + h]);
    scores[tid] = ex;
    atomicAdd(&denom[(size_t)d * 8 + h], ex);
}

// ---------------------------------------------------------------------------
// 8) Pass 3: ve = rel_e @ Wve (fused WMMA); msg = (vn+ve)*alpha;
//    agg[dst] += msg  (segment sum via f32 atomics).
// ---------------------------------------------------------------------------
__global__ void attn_agg_kernel(const unsigned short* __restrict__ relE,
                                const unsigned short* __restrict__ Wvep,
                                const float* __restrict__ bve,
                                const float* __restrict__ Vn,
                                const float* __restrict__ scores,
                                const float* __restrict__ denom,
                                const int* __restrict__ eidx,
                                float* __restrict__ agg)
{
    int lane = threadIdx.x & 31;
    int wid = (blockIdx.x * blockDim.x + threadIdx.x) >> 5;
    int nw = (gridDim.x * blockDim.x) >> 5;
    int half = (lane >> 4) & 1;

    for (int t = wid; t < (EE >> 4); t += nw) {
        v16bf a0 = load_a_bf16(relE, (long)t * 16, 64, 0, lane);
        v16bf a1 = load_a_bf16(relE, (long)t * 16, 64, 1, lane);
        int es[8], ds[8];
#pragma unroll
        for (int rr = 0; rr < 8; ++rr) {
            int e = t * 16 + rr + 8 * half;
            es[rr] = eidx[e]; ds[rr] = eidx[EE + e];
        }
#pragma unroll
        for (int nt = 0; nt < 4; ++nt) {
            v8f acc = {};
            acc = wmma_bf16(a0, load_b_packed(Wvep, 4, nt, 0, lane), acc);
            acc = wmma_bf16(a1, load_b_packed(Wvep, 4, nt, 1, lane), acc);
            int n = nt * 16 + (lane & 15);
            float bb = bve[n];
            int h = nt * 2 + ((lane >> 3) & 1);
#pragma unroll
            for (int rr = 0; rr < 8; ++rr) {
                int e = t * 16 + rr + 8 * half;
                float al = scores[(size_t)e * 8 + h] /
                           (denom[(size_t)ds[rr] * 8 + h] + 1e-16f);
                float vn = Vn[(size_t)es[rr] * 64 + n];
                atomicAdd(&agg[(size_t)ds[rr] * 64 + n], (vn + acc[rr] + bb) * al);
            }
        }
    }
}

// ---------------------------------------------------------------------------
// 9) GRU-style node update: gate = sigmoid(agg@Wih + xn@Whh + b);
//    upd = agg + gate*(xn@Wself + bself - agg); x += upd@Wout + bout.
//    4 fused WMMA GEMMs per tile; upd bounced through LDS to re-fragment.
// ---------------------------------------------------------------------------
__global__ void node_update_kernel(const float* __restrict__ aggp,
                                   const unsigned short* __restrict__ xn,
                                   const unsigned short* __restrict__ Wihp, const float* __restrict__ bih,
                                   const unsigned short* __restrict__ Whhp, const float* __restrict__ bhh,
                                   const unsigned short* __restrict__ Wselfp, const float* __restrict__ bself,
                                   const unsigned short* __restrict__ Woutp, const float* __restrict__ bout,
                                   float* __restrict__ x)
{
    __shared__ unsigned short uts[4][16][72];
    int lane = threadIdx.x & 31;
    int wv = threadIdx.x >> 5;
    int wid = (blockIdx.x * blockDim.x + threadIdx.x) >> 5;
    int nw = (gridDim.x * blockDim.x) >> 5;
    int half = (lane >> 4) & 1;

    for (int t = wid; t < (NN >> 4); t += nw) {
        v16bf aa0 = load_a_f32(aggp, (long)t * 16, 64, 0, lane);
        v16bf aa1 = load_a_f32(aggp, (long)t * 16, 64, 1, lane);
        v16bf ax0 = load_a_bf16(xn, (long)t * 16, 64, 0, lane);
        v16bf ax1 = load_a_bf16(xn, (long)t * 16, 64, 1, lane);
#pragma unroll
        for (int nt = 0; nt < 4; ++nt) {
            v8f g1 = {}, g2 = {}, sv = {};
            g1 = wmma_bf16(aa0, load_b_packed(Wihp, 4, nt, 0, lane), g1);
            g1 = wmma_bf16(aa1, load_b_packed(Wihp, 4, nt, 1, lane), g1);
            g2 = wmma_bf16(ax0, load_b_packed(Whhp, 4, nt, 0, lane), g2);
            g2 = wmma_bf16(ax1, load_b_packed(Whhp, 4, nt, 1, lane), g2);
            sv = wmma_bf16(ax0, load_b_packed(Wselfp, 4, nt, 0, lane), sv);
            sv = wmma_bf16(ax1, load_b_packed(Wselfp, 4, nt, 1, lane), sv);
            int n = nt * 16 + (lane & 15);
            float b12 = bih[n] + bhh[n], bs = bself[n];
#pragma unroll
            for (int rr = 0; rr < 8; ++rr) {
                int row = t * 16 + rr + 8 * half;
                float av = aggp[(size_t)row * 64 + n];
                float gt = 1.f / (1.f + __expf(-(g1[rr] + g2[rr] + b12)));
                float ud = av + gt * ((sv[rr] + bs) - av);
                uts[wv][rr + 8 * half][n] = f2bf(ud);
            }
        }
        wait_ds();
        v16bf u0 = load_a_bf16(&uts[wv][0][0], 0, 72, 0, lane);
        v16bf u1 = load_a_bf16(&uts[wv][0][0], 0, 72, 1, lane);
#pragma unroll
        for (int nt = 0; nt < 4; ++nt) {
            int n = nt * 16 + (lane & 15);
            float bo = bout[n];
            v8f acc;
#pragma unroll
            for (int rr = 0; rr < 8; ++rr)
                acc[rr] = x[(size_t)(t * 16 + rr + 8 * half) * 64 + n] + bo;
            acc = wmma_bf16(u0, load_b_packed(Woutp, 4, nt, 0, lane), acc);
            acc = wmma_bf16(u1, load_b_packed(Woutp, 4, nt, 1, lane), acc);
#pragma unroll
            for (int rr = 0; rr < 8; ++rr)
                x[(size_t)(t * 16 + rr + 8 * half) * 64 + n] = acc[rr];
        }
        wait_ds();
    }
}

// ---------------------------------------------------------------------------
// 10) FFN: x += relu(xn2 @ mw1 + mb1) @ mw2 + mb2.  64->256->64, WMMA both,
//     intermediate ReLU tile staged in LDS (bf16), residual in accumulator.
// ---------------------------------------------------------------------------
__global__ void ffn_kernel(const unsigned short* __restrict__ xn2,
                           const unsigned short* __restrict__ mw1p, const float* __restrict__ mb1,
                           const unsigned short* __restrict__ mw2p, const float* __restrict__ mb2,
                           float* __restrict__ x)
{
    __shared__ unsigned short tls[4][16][264];
    int lane = threadIdx.x & 31;
    int wv = threadIdx.x >> 5;
    int wid = (blockIdx.x * blockDim.x + threadIdx.x) >> 5;
    int nw = (gridDim.x * blockDim.x) >> 5;
    int half = (lane >> 4) & 1;

    for (int t = wid; t < (NN >> 4); t += nw) {
        v16bf a0 = load_a_bf16(xn2, (long)t * 16, 64, 0, lane);
        v16bf a1 = load_a_bf16(xn2, (long)t * 16, 64, 1, lane);
#pragma unroll
        for (int nt = 0; nt < 16; ++nt) {
            v8f acc = {};
            acc = wmma_bf16(a0, load_b_packed(mw1p, 16, nt, 0, lane), acc);
            acc = wmma_bf16(a1, load_b_packed(mw1p, 16, nt, 1, lane), acc);
            int n = nt * 16 + (lane & 15);
            float bb = mb1[n];
#pragma unroll
            for (int rr = 0; rr < 8; ++rr)
                tls[wv][rr + 8 * half][n] = f2bf(fmaxf(acc[rr] + bb, 0.f));
        }
        wait_ds();
#pragma unroll
        for (int nt = 0; nt < 4; ++nt) {
            int n = nt * 16 + (lane & 15);
            float bo = mb2[n];
            v8f acc;
#pragma unroll
            for (int rr = 0; rr < 8; ++rr)
                acc[rr] = x[(size_t)(t * 16 + rr + 8 * half) * 64 + n] + bo;
#pragma unroll
            for (int kb = 0; kb < 8; ++kb) {
                v16bf ta = load_a_bf16(&tls[wv][0][0], 0, 264, kb, lane);
                acc = wmma_bf16(ta, load_b_packed(mw2p, 4, nt, kb, lane), acc);
            }
#pragma unroll
            for (int rr = 0; rr < 8; ++rr)
                x[(size_t)(t * 16 + rr + 8 * half) * 64 + n] = acc[rr];
        }
        wait_ds();
    }
}

// ---------------------------------------------------------------------------
// Host orchestration. Input order = setup_inputs() dict insertion order,
// nested dicts flattened in insertion order.
// ---------------------------------------------------------------------------
extern "C" void kernel_launch(void* const* d_in, const int* in_sizes, int n_in,
                              void* d_out, int out_size, void* d_ws, size_t ws_size,
                              hipStream_t stream) {
    const float* positions = (const float*)d_in[0];
    const unsigned char* pad = (const unsigned char*)d_in[1];
    const float* ang = (const float*)d_in[2];
    const int* eidx = (const int*)d_in[3];
    const float* local_embed = (const float*)d_in[4];
    // d_in[5] nearby_mask unused by the reference math
    const float* r1w0 = (const float*)d_in[6];
    const float* r1b0 = (const float*)d_in[7];
    const float* r1g0 = (const float*)d_in[8];
    const float* r1bb0 = (const float*)d_in[9];
    const float* r1w1 = (const float*)d_in[10];
    const float* r1b1 = (const float*)d_in[11];
    const float* r2w0 = (const float*)d_in[12];
    const float* r2b0 = (const float*)d_in[13];
    const float* r2g0 = (const float*)d_in[14];
    const float* r2bb0 = (const float*)d_in[15];
    const float* r2w1 = (const float*)d_in[16];
    const float* r2b1 = (const float*)d_in[17];
    const float* a_g0 = (const float*)d_in[18];
    const float* a_b0 = (const float*)d_in[19];
    const float* a_w = (const float*)d_in[20];
    const float* a_b = (const float*)d_in[21];
    const float* a_g1 = (const float*)d_in[22];
    const float* a_b1 = (const float*)d_in[23];
    const float* Wq = (const float*)d_in[24];   const float* bq = (const float*)d_in[25];
    const float* Wkn = (const float*)d_in[26];  const float* bkn = (const float*)d_in[27];
    const float* Wke = (const float*)d_in[28];  const float* bke = (const float*)d_in[29];
    const float* Wvn = (const float*)d_in[30];  const float* bvn = (const float*)d_in[31];
    const float* Wve = (const float*)d_in[32];  const float* bve = (const float*)d_in[33];
    const float* Wself = (const float*)d_in[34];const float* bself = (const float*)d_in[35];
    const float* Wih = (const float*)d_in[36];  const float* bih = (const float*)d_in[37];
    const float* Whh = (const float*)d_in[38];  const float* bhh = (const float*)d_in[39];
    const float* Wout = (const float*)d_in[40]; const float* bout = (const float*)d_in[41];
    const float* n1g = (const float*)d_in[42];  const float* n1b = (const float*)d_in[43];
    const float* n2g = (const float*)d_in[44];  const float* n2b = (const float*)d_in[45];
    const float* mw1 = (const float*)d_in[46];  const float* mb1 = (const float*)d_in[47];
    const float* mw2 = (const float*)d_in[48];  const float* mb2 = (const float*)d_in[49];

    float* x = (float*)d_out;

    // workspace carve (256B aligned)
    char* wp = (char*)d_ws;
    auto carve = [&](size_t bytes) {
        void* p = (void*)wp;
        wp += (bytes + 255) & ~(size_t)255;
        return p;
    };
    unsigned short* relE = (unsigned short*)carve((size_t)EE * 64 * 2);  // bf16 [E,64]
    unsigned short* ubuf = (unsigned short*)carve((size_t)EE * 64 * 2);  // bf16 [E,64]
    float* scores = (float*)ubuf;  // alias: ubuf dead after gemm_ln; scores = [E,8] f32
    unsigned short* xn  = (unsigned short*)carve((size_t)NN * 64 * 2);
    unsigned short* xn2 = (unsigned short*)carve((size_t)NN * 64 * 2);
    float* Qn = (float*)carve((size_t)NN * 64 * 4);
    float* Kn = (float*)carve((size_t)NN * 64 * 4);
    float* Vn = (float*)carve((size_t)NN * 64 * 4);
    float* mbuf = (float*)carve((size_t)NN * 8 * 4);
    float* denom = (float*)carve((size_t)NN * 8 * 4);
    float* agg = (float*)carve((size_t)NN * 64 * 4);

    // packed weight arena (bf16 fragment order)
    const size_t PK64 = 4096;       // 64x64 -> 8 blocks * 32 lanes * 16
    const size_t PKBIG = 16384;     // 64x256 or 256x64 -> 32 blocks
    unsigned short* aw_p = (unsigned short*)carve(PK64 * 2);
    unsigned short* wp9[LL][9];
    unsigned short* mw1_p[LL];
    unsigned short* mw2_p[LL];
    for (int l = 0; l < LL; ++l) {
        for (int i = 0; i < 9; ++i) wp9[l][i] = (unsigned short*)carve(PK64 * 2);
        mw1_p[l] = (unsigned short*)carve(PKBIG * 2);
        mw2_p[l] = (unsigned short*)carve(PKBIG * 2);
    }

    // x <- local_embed
    hipMemcpyAsync(x, local_embed, (size_t)NN * 64 * 4, hipMemcpyDeviceToDevice, stream);

    // weight pre-pack (tiny, once per call)
    prepack_b_kernel<<<2, 128, 0, stream>>>(a_w, 64, 64, aw_p);
    const float* w64[9] = {Wq, Wkn, Wke, Wvn, Wve, Wself, Wih, Whh, Wout};
    for (int l = 0; l < LL; ++l) {
        for (int i = 0; i < 9; ++i)
            prepack_b_kernel<<<2, 128, 0, stream>>>(w64[i] + (size_t)l * 64 * 64, 64, 64, wp9[l][i]);
        prepack_b_kernel<<<8, 128, 0, stream>>>(mw1 + (size_t)l * 64 * 256, 64, 256, mw1_p[l]);
        prepack_b_kernel<<<8, 128, 0, stream>>>(mw2 + (size_t)l * 256 * 64, 256, 64, mw2_p[l]);
    }

    // edge embedding
    rel_phase1_kernel<<<EE / 256, 256, 0, stream>>>(
        positions, ang, eidx,
        r1w0, r1b0, r1g0, r1bb0, r1w1, r1b1,
        r2w0, r2b0, r2g0, r2bb0, r2w1, r2b1,
        a_g0, a_b0, ubuf);
    gemm_ln_kernel<<<1024, 128, 0, stream>>>(ubuf, aw_p, a_b, a_g1, a_b1, relE, EE);

    for (int l = 0; l < LL; ++l) {
        size_t bO = (size_t)l * 64;
        const float* b64[9] = {bq + bO, bkn + bO, bke + bO, bvn + bO, bve + bO,
                               bself + bO, bih + bO, bhh + bO, bout + bO};

        ln64_kernel<<<NN / 256, 256, 0, stream>>>(x, n1g + bO, n1b + bO, xn, NN);
        qkv_kernel<<<256, 128, 0, stream>>>(xn, wp9[l][0], b64[0], wp9[l][1], b64[1],
                                            wp9[l][3], b64[3], Qn, Kn, Vn);
        init_kernel<<<(NN * DD) / 256, 256, 0, stream>>>(mbuf, denom, agg);
        attn_score_kernel<<<2048, 128, 0, stream>>>(relE, wp9[l][2], b64[2],
                                                    Qn, Kn, eidx, pad, scores, mbuf);
        attn_exp_kernel<<<(EE * 8) / 256, 256, 0, stream>>>(scores, mbuf, denom, eidx);
        attn_agg_kernel<<<2048, 128, 0, stream>>>(relE, wp9[l][4], b64[4],
                                                  Vn, scores, denom, eidx, agg);
        node_update_kernel<<<256, 128, 0, stream>>>(agg, xn,
                                                    wp9[l][6], b64[6], wp9[l][7], b64[7],
                                                    wp9[l][5], b64[5], wp9[l][8], b64[8],
                                                    x);
        ln64_kernel<<<NN / 256, 256, 0, stream>>>(x, n2g + bO, n2b + bO, xn2, NN);
        ffn_kernel<<<256, 128, 0, stream>>>(xn2, mw1_p[l], mb1 + (size_t)l * 256,
                                            mw2_p[l], mb2 + bO, x);
    }
}
